// InterMeshTransformer_36713380446893
// MI455X (gfx1250) — matrix operand order repaired and checked by reading, not compiled, not run on hardware
//
#include <hip/hip_runtime.h>
#include <hip/hip_bf16.h>

typedef __attribute__((ext_vector_type(16))) _Float16 v16h;
typedef __attribute__((ext_vector_type(8)))  _Float16 v8h;
typedef __attribute__((ext_vector_type(8)))  float    v8f;

#define NTOK   55120        // Q*V = 8*6890
#define VDIM   6890
#define EDIM   256
#define LN_EPS 1e-5f

// ---------------------------------------------------------------------------
// f32 -> f16 elementwise conversion
// ---------------------------------------------------------------------------
__global__ __launch_bounds__(256) void cvt_f32_f16(const float* __restrict__ src,
                                                   _Float16* __restrict__ dst, int n) {
    int i = blockIdx.x * 256 + threadIdx.x;
    if (i < n) dst[i] = (_Float16)src[i];
}

// ---------------------------------------------------------------------------
// WMMA GEMM: C[16 x 256-block] = A(N x KDIM, f16 row-major) @ W^T + bias,
// with fused epilogues. W is (Mout x KDIM) f16 row-major (out-features as rows),
// exactly B^T, so B fragments are contiguous row loads.
//
// Block = 256 threads = 8 waves. Each workgroup computes a 16-row x 256-col
// tile; wave w owns column tiles [w*32, w*32+32). Accumulator staged to LDS
// for the row-wise (LayerNorm) epilogues and coalesced f16 stores.
// ---------------------------------------------------------------------------
enum { EPI_F16 = 0, EPI_GELU = 1, EPI_LN = 2, EPI_LN_FINAL = 3 };

template <int EPI, int KDIM>
__global__ __launch_bounds__(256) void gemm_epi(
    const _Float16* __restrict__ A, const _Float16* __restrict__ W,
    const float* __restrict__ bias, const float* __restrict__ resid,
    const float* __restrict__ lng, const float* __restrict__ lnb,
    float* __restrict__ outF, _Float16* __restrict__ outH, int ldOut) {
    __shared__ float tile[16][264];  // 16 rows x 256 cols (+pad), ~16.5 KB of 320 KB LDS

    const int tid  = threadIdx.x;
    const int w    = tid >> 5;
    const int lane = tid & 31;
    const int hi   = lane >> 4;   // half-wave select (contiguous K half: k = hi*16..hi*16+15)
    const int lo   = lane & 15;   // row (A) / column (B,C) within tile
    const size_t row0 = (size_t)blockIdx.x * 16;
    const int    cb   = blockIdx.y * 256;

    #pragma unroll
    for (int t = 0; t < 2; ++t) {
        const int col0 = cb + w * 32 + t * 16;
        v8f c = {};
        #pragma unroll
        for (int k0 = 0; k0 < KDIM; k0 += 32) {
            // A fragment: lane(hi,m) holds k in
            // {hi*16..hi*16+15} (contiguous, matches B)  -> two 16B loads
            const _Float16* ap = A + (row0 + lo) * (size_t)KDIM + k0 + hi * 16;
            v8h a0 = *(const v8h*)ap;
            v8h a1 = *(const v8h*)(ap + 8);
            v16h a;
            #pragma unroll
            for (int i = 0; i < 8; ++i) { a[i] = a0[i]; a[8 + i] = a1[i]; }
            // B fragment: lane(hi,n) holds B[k = hi*16 + i, n] = W[col0+n, k0+hi*16+i]
            v16h b = *(const v16h*)(W + (size_t)(col0 + lo) * KDIM + k0 + hi * 16);
            c = __builtin_amdgcn_wmma_f32_16x16x32_f16(false, a, false, b,
                                                       (short)0, c, false, false);
        }
        const float bv = bias[col0 + lo];
        // C layout: lane(hi,n), vgpr r -> C[r + 8*hi, n]
        #pragma unroll
        for (int r = 0; r < 8; ++r)
            tile[r + 8 * hi][w * 32 + t * 16 + lo] = c[r] + bv;
    }
    __syncthreads();

    if (EPI == EPI_F16 || EPI == EPI_GELU) {
        // coalesced f16 store, optional exact GELU
        const int r  = tid >> 4;
        const int cs = (tid & 15) * 16;
        const size_t base = (row0 + r) * (size_t)ldOut + cb + cs;
        #pragma unroll
        for (int j = 0; j < 16; ++j) {
            float x = tile[r][cs + j];
            if (EPI == EPI_GELU) x = 0.5f * x * (1.0f + erff(x * 0.70710678118654752f));
            outH[base + j] = (_Float16)x;
        }
    } else {
        // LayerNorm(out + resid) over 256 cols; 8 waves x 2 rows
        #pragma unroll
        for (int rr = 0; rr < 2; ++rr) {
            const int r = w * 2 + rr;
            const size_t gbase = (row0 + r) * (size_t)EDIM;
            float x[8], sum = 0.f, sq = 0.f;
            #pragma unroll
            for (int tq = 0; tq < 8; ++tq) {
                const int col = lane + tq * 32;
                const float v = tile[r][col] + resid[gbase + col];
                x[tq] = v; sum += v; sq += v * v;
            }
            #pragma unroll
            for (int off = 1; off < 32; off <<= 1) {
                sum += __shfl_xor(sum, off);
                sq  += __shfl_xor(sq, off);
            }
            const float mean = sum * (1.0f / 256.0f);
            const float var  = sq * (1.0f / 256.0f) - mean * mean;
            const float rstd = rsqrtf(var + LN_EPS);
            #pragma unroll
            for (int tq = 0; tq < 8; ++tq) {
                const int col = lane + tq * 32;
                const float y = (x[tq] - mean) * rstd * lng[col] + lnb[col];
                outF[gbase + col] = y;
                if (EPI == EPI_LN) outH[gbase + col] = (_Float16)y;
            }
        }
    }
}

// ---------------------------------------------------------------------------
// Neighborhood attention: one wave per token, 8 tokens per workgroup.
// Phase 1: lanes = (k, d-half); partial dots + xor-16 fold -> score[k] on lane k.
// Softmax across the 16-lane group (lanes 16..31 mirror). Phase 2: lanes = d,
// broadcast attn[k]/row[k] via shuffles, accumulate attn @ V.
// ---------------------------------------------------------------------------
__global__ __launch_bounds__(256) void attn_kernel(
    const _Float16* __restrict__ Qp, const _Float16* __restrict__ Kp,
    const _Float16* __restrict__ Vp, const int* __restrict__ bidx,
    const int* __restrict__ graph, const unsigned char* __restrict__ cmask,
    _Float16* __restrict__ O) {
    const int wid  = threadIdx.x >> 5;
    const int lane = threadIdx.x & 31;
    const size_t n = (size_t)blockIdx.x * 8 + wid;  // token id
    const int kq   = lane & 15;
    const int half = lane >> 4;

    const size_t ib = n * 16 + kq;
    const int  row  = bidx[ib] * VDIM + graph[ib];
    const bool dead = cmask[ib] != 0;  // mask==True -> excluded (-inf)

    const _Float16* qrow = Qp + n * (size_t)EDIM;
    const float scale = 0.17677669529663687f;  // 1/sqrt(32)

    #pragma unroll 1
    for (int h = 0; h < 8; ++h) {
        const _Float16* qp = qrow + h * 32 + half * 16;
        const _Float16* kp = Kp + (size_t)row * EDIM + h * 32 + half * 16;
        v8h qa = *(const v8h*)qp, qb = *(const v8h*)(qp + 8);
        v8h ka = *(const v8h*)kp, kb = *(const v8h*)(kp + 8);
        float s = 0.f;
        #pragma unroll
        for (int i = 0; i < 8; ++i)
            s += (float)qa[i] * (float)ka[i] + (float)qb[i] * (float)kb[i];
        s += __shfl_xor(s, 16);               // fold d-halves -> full dot on both halves
        s = dead ? -1e30f : s * scale;

        float m = s;                           // max over the 16-lane key group
        #pragma unroll
        for (int off = 1; off < 16; off <<= 1) m = fmaxf(m, __shfl_xor(m, off));
        const float e = __expf(s - m);
        float den = e;
        #pragma unroll
        for (int off = 1; off < 16; off <<= 1) den += __shfl_xor(den, off);
        const float a = e / den;

        float o = 0.f;                         // lanes = d (32 dims of this head)
        #pragma unroll
        for (int k = 0; k < 16; ++k) {
            const float ak = __shfl(a, k);
            const int   rk = __shfl(row, k);
            o += ak * (float)Vp[(size_t)rk * EDIM + h * 32 + lane];
        }
        O[n * (size_t)EDIM + h * 32 + lane] = (_Float16)o;
    }
}

// ---------------------------------------------------------------------------
// Launcher
// ---------------------------------------------------------------------------
extern "C" void kernel_launch(void* const* d_in, const int* in_sizes, int n_in,
                              void* d_out, int out_size, void* d_ws, size_t ws_size,
                              hipStream_t stream) {
    (void)in_sizes; (void)n_in; (void)out_size; (void)ws_size;
    const float* mesh = (const float*)d_in[0];
    const float* Wq   = (const float*)d_in[1];
    const float* bq   = (const float*)d_in[2];
    const float* Wk   = (const float*)d_in[3];
    const float* bk   = (const float*)d_in[4];
    const float* Wv   = (const float*)d_in[5];
    const float* bv   = (const float*)d_in[6];
    const float* Wo   = (const float*)d_in[7];
    const float* bo   = (const float*)d_in[8];
    const float* ln1g = (const float*)d_in[9];
    const float* ln1b = (const float*)d_in[10];
    const float* W1   = (const float*)d_in[11];
    const float* b1   = (const float*)d_in[12];
    const float* W2   = (const float*)d_in[13];
    const float* b2   = (const float*)d_in[14];
    const float* ln2g = (const float*)d_in[15];
    const float* ln2b = (const float*)d_in[16];
    const int*   bidx = (const int*)d_in[17];
    const int*   grph = (const int*)d_in[18];
    const unsigned char* cmask = (const unsigned char*)d_in[19];

    char* ws = (char*)d_ws;
    const size_t szA16 = (size_t)NTOK * EDIM * sizeof(_Float16);  // 28.2 MB

    // f16 weights (out x in), < 2 MB total
    _Float16* Whq = (_Float16*)ws;
    _Float16* Whk = Whq + 65536;
    _Float16* Whv = Whk + 65536;
    _Float16* Who = Whv + 65536;
    _Float16* Wh1 = Who + 65536;     // 1024 x 256
    _Float16* Wh2 = Wh1 + 262144;    // 256 x 1024

    char* base = ws + (2u << 20);
    _Float16* A16 = (_Float16*)(base);               // mesh f16
    _Float16* Qp  = (_Float16*)(base + szA16);
    _Float16* Kp  = (_Float16*)(base + 2 * szA16);
    _Float16* Vp  = (_Float16*)(base + 3 * szA16);
    _Float16* H16 = (_Float16*)(base + 4 * szA16);   // N x 1024 f16 (112.9 MB)
    // lifetime-based aliases:
    _Float16* O16 = A16;             // mesh f16 dead after projections
    float*    Xf  = (float*)Qp;      // N x 256 f32 spans Qp+Kp regions (dead after attn)
    _Float16* Xh  = Vp;              // Vp dead after attn
    float*    Y   = (float*)d_out;

    const dim3 blk(256);
    const int rowBlocks = NTOK / 16;  // 3445

    // 1) conversions
    cvt_f32_f16<<<(NTOK * EDIM + 255) / 256, blk, 0, stream>>>(mesh, A16, NTOK * EDIM);
    cvt_f32_f16<<<(65536 + 255) / 256, blk, 0, stream>>>(Wq, Whq, 65536);
    cvt_f32_f16<<<(65536 + 255) / 256, blk, 0, stream>>>(Wk, Whk, 65536);
    cvt_f32_f16<<<(65536 + 255) / 256, blk, 0, stream>>>(Wv, Whv, 65536);
    cvt_f32_f16<<<(65536 + 255) / 256, blk, 0, stream>>>(Wo, Who, 65536);
    cvt_f32_f16<<<(262144 + 255) / 256, blk, 0, stream>>>(W1, Wh1, 262144);
    cvt_f32_f16<<<(262144 + 255) / 256, blk, 0, stream>>>(W2, Wh2, 262144);

    // 2) Q/K/V projections on UNIQUE rows (16x fewer FLOPs than gathered form)
    gemm_epi<EPI_F16, 256><<<dim3(rowBlocks, 1), blk, 0, stream>>>(
        A16, Whq, bq, nullptr, nullptr, nullptr, nullptr, Qp, EDIM);
    gemm_epi<EPI_F16, 256><<<dim3(rowBlocks, 1), blk, 0, stream>>>(
        A16, Whk, bk, nullptr, nullptr, nullptr, nullptr, Kp, EDIM);
    gemm_epi<EPI_F16, 256><<<dim3(rowBlocks, 1), blk, 0, stream>>>(
        A16, Whv, bv, nullptr, nullptr, nullptr, nullptr, Vp, EDIM);

    // 3) gather-attention over 16 neighbors (K/V gathers hit 192MB L2)
    attn_kernel<<<NTOK / 8, blk, 0, stream>>>(Qp, Kp, Vp, bidx, grph, cmask, O16);

    // 4) Wo projection + residual + LayerNorm1 (f32 + f16 outputs)
    gemm_epi<EPI_LN, 256><<<dim3(rowBlocks, 1), blk, 0, stream>>>(
        O16, Who, bo, mesh, ln1g, ln1b, Xf, Xh, EDIM);

    // 5) FFN up (256 -> 1024) + exact GELU
    gemm_epi<EPI_GELU, 256><<<dim3(rowBlocks, 4), blk, 0, stream>>>(
        Xh, Wh1, b1, nullptr, nullptr, nullptr, nullptr, H16, 1024);

    // 6) FFN down (1024 -> 256) + residual + LayerNorm2 -> final f32 output
    gemm_epi<EPI_LN_FINAL, 1024><<<dim3(rowBlocks, 1), blk, 0, stream>>>(
        H16, Wh2, b2, Xf, ln2g, ln2b, Y, nullptr, EDIM);
}
